// GCN2_3650722202372
// MI455X (gfx1250) — compile-verified, matrix operand order checked by reading, not run on hardware
//
#include <hip/hip_runtime.h>
#include <hip/hip_bf16.h>

typedef __attribute__((ext_vector_type(16))) _Float16 v16h;
typedef __attribute__((ext_vector_type(8)))  _Float16 v8h;
typedef __attribute__((ext_vector_type(8)))  float    v8f;

// ---------------------------------------------------------------------------
// Degree / normalization kernels
// ---------------------------------------------------------------------------
__global__ void k_deg_init(float* __restrict__ deg, int N) {
    int i = blockIdx.x * blockDim.x + threadIdx.x;
    if (i < N) deg[i] = 1.0f;                       // self-loop contribution
}

__global__ void k_deg_count(const int* __restrict__ col, float* __restrict__ deg, int E) {
    int e = blockIdx.x * blockDim.x + threadIdx.x;
    if (e < E)
        __hip_atomic_fetch_add(&deg[col[e]], 1.0f,
                               __ATOMIC_RELAXED, __HIP_MEMORY_SCOPE_AGENT);
}

__global__ void k_deg_rsqrt(float* __restrict__ deg, int N) {
    int i = blockIdx.x * blockDim.x + threadIdx.x;
    if (i < N) {
        float d = deg[i];
        deg[i] = (d > 0.0f) ? rsqrtf(d) : 0.0f;     // deg^{-1/2}
    }
}

// ---------------------------------------------------------------------------
// One-time repack of W (K x NOUT, f32 row-major) into the per-lane WMMA
// B-fragment layout as f16:
//   Wp[(((ntile*4 + ks)*32 + lane)*16) + j]
// where j<8  -> W[ks*32 + hi*8 + j      ][ntile*16 + l16]
//       j>=8 -> W[ks*32 + hi*8 + 16 + (j-8)][ntile*16 + l16]
// Each wave then fetches its B fragment as one contiguous 32B v16h load.
// ---------------------------------------------------------------------------
template <int NOUT>
__global__ void k_packW(const float* __restrict__ W, _Float16* __restrict__ Wp) {
    constexpr int K = 128;
    constexpr int total = K * NOUT;
    int tid = blockIdx.x * blockDim.x + threadIdx.x;
    if (tid >= total) return;
    int j     = tid & 15;
    int lane  = (tid >> 4) & 31;
    int ks    = (tid >> 9) & 3;
    int ntile = tid >> 11;
    int hi    = lane >> 4;
    int l16   = lane & 15;
    int koff  = ks * 32 + hi * 8;
    int krow  = koff + ((j < 8) ? j : (j + 8));     // +16 gap for the upper 8 halfs
    int n     = ntile * 16 + l16;
    Wp[tid] = (_Float16)W[(size_t)krow * NOUT + n];
}

// ---------------------------------------------------------------------------
// WMMA GEMM:  C[M x NOUT] = A[M x 128] @ W[128 x NOUT]
//  - 16-row stripe of A staged through LDS as f16 (shared by all waves)
//  - B fragments come pre-packed from Wp (one v16h load per K-step)
//  - v_wmma_f32_16x16x32_f16, f32 accumulate; one wave -> one 16x16 tile
// ---------------------------------------------------------------------------
template <int NOUT>
__global__ __launch_bounds__(32 * (NOUT / 16))
void k_gemm_wmma(const float* __restrict__ A, const _Float16* __restrict__ Wp,
                 float* __restrict__ C, int M) {
    constexpr int K    = 128;
    constexpr int LDSK = K + 8;                    // pad: 272B row stride kills bank conflicts
    __shared__ _Float16 As[16 * LDSK];

    const int tid = threadIdx.x;
    const int m0  = blockIdx.x * 16;

    // Cooperative fill: 16 rows x 128 cols f32 -> f16 in LDS (8 elems/thread/iter)
    for (int idx = tid * 8; idx < 16 * K; idx += blockDim.x * 8) {
        int rrow = idx >> 7;
        int ccol = idx & (K - 1);
        int mg = m0 + rrow;
        if (mg >= M) mg = M - 1;                   // clamp (M is a multiple of 16 here)
        const float4* src = (const float4*)(A + (size_t)mg * K + ccol);
        float4 f0 = src[0];
        float4 f1 = src[1];
        _Float16* dst = &As[rrow * LDSK + ccol];
        dst[0] = (_Float16)f0.x; dst[1] = (_Float16)f0.y;
        dst[2] = (_Float16)f0.z; dst[3] = (_Float16)f0.w;
        dst[4] = (_Float16)f1.x; dst[5] = (_Float16)f1.y;
        dst[6] = (_Float16)f1.z; dst[7] = (_Float16)f1.w;
    }
    __syncthreads();

    const int lane = tid & 31;
    const int wave = tid >> 5;                     // column-tile index
    const int l16  = lane & 15;
    const int hi   = lane >> 4;

    v8f acc = {};
#pragma unroll
    for (int ks = 0; ks < 4; ++ks) {
        const int koff = ks * 32 + hi * 8;
        // A fragment: two 16B LDS reads (ds_load_b128)
        const _Float16* ap = &As[l16 * LDSK + koff];
        v8h alo = *(const v8h*)ap;
        v8h ahi = *(const v8h*)(ap + 16);
        v16h av;
#pragma unroll
        for (int j = 0; j < 8; ++j) { av[j] = alo[j]; av[8 + j] = ahi[j]; }

        // B fragment: one contiguous 32B load from the packed weights
        v16h bv = *(const v16h*)(Wp + ((((wave * 4 + ks) * 32) + lane) << 4));

        acc = __builtin_amdgcn_wmma_f32_16x16x32_f16(
            /*neg_a=*/false, av, /*neg_b=*/false, bv,
            /*c_mod=*/(short)0, acc, /*reuse_a=*/false, /*reuse_b=*/false);
    }

    // C/D layout: VGPR v -> row v + 8*hi, lane%16 -> column
    const int n = wave * 16 + l16;
#pragma unroll
    for (int v = 0; v < 8; ++v) {
        int mr = m0 + v + hi * 8;
        if (mr < M) C[(size_t)mr * NOUT + n] = acc[v];
    }
}

// ---------------------------------------------------------------------------
// Self-loop term (also initializes the accumulation buffer):
//   out[i][c] = t[i][c] * dis[i]^2
// ---------------------------------------------------------------------------
template <int CH>
__global__ void k_self(const float* __restrict__ t, const float* __restrict__ dis,
                       float* __restrict__ out, int N) {
    constexpr int Q = CH / 4;                      // float4s per node
    size_t i = (size_t)blockIdx.x * blockDim.x + threadIdx.x;
    size_t total = (size_t)N * Q;
    if (i >= total) return;
    size_t node = i / Q;
    float s = dis[node]; s = s * s;
    float4 v = ((const float4*)t)[i];
    float4 o; o.x = v.x * s; o.y = v.y * s; o.z = v.z * s; o.w = v.w * s;
    ((float4*)out)[i] = o;
}

// ---------------------------------------------------------------------------
// Edge scatter: one wave per edge, lane covers CPL consecutive channels.
//   dst[col] += src[row] * dis[row] * dis[col]
// CPL = 4 (C=128) or 2 (C=64).  Bandwidth-bound: ~2.5GB total -> ~110us @ 23.3TB/s.
// ---------------------------------------------------------------------------
template <int CPL>
__global__ void k_edges(const float* __restrict__ src,
                        const int* __restrict__ row, const int* __restrict__ col,
                        const float* __restrict__ dis,
                        float* __restrict__ dst, int E) {
    constexpr int C = CPL * 32;
    long long gid = (long long)blockIdx.x * blockDim.x + threadIdx.x;
    int e    = (int)(gid >> 5);
    int lane = (int)(gid & 31);
    if (e >= E) return;
    int r = row[e];
    int c = col[e];
    float nrm = dis[r] * dis[c];
    const float* __restrict__ s = src + (size_t)r * C + lane * CPL;
    float* __restrict__ d       = dst + (size_t)c * C + lane * CPL;
    if (CPL == 4) {
        float4 v = *(const float4*)s;
        __hip_atomic_fetch_add(d + 0, v.x * nrm, __ATOMIC_RELAXED, __HIP_MEMORY_SCOPE_AGENT);
        __hip_atomic_fetch_add(d + 1, v.y * nrm, __ATOMIC_RELAXED, __HIP_MEMORY_SCOPE_AGENT);
        __hip_atomic_fetch_add(d + 2, v.z * nrm, __ATOMIC_RELAXED, __HIP_MEMORY_SCOPE_AGENT);
        __hip_atomic_fetch_add(d + 3, v.w * nrm, __ATOMIC_RELAXED, __HIP_MEMORY_SCOPE_AGENT);
    } else {
        float2 v = *(const float2*)s;
        __hip_atomic_fetch_add(d + 0, v.x * nrm, __ATOMIC_RELAXED, __HIP_MEMORY_SCOPE_AGENT);
        __hip_atomic_fetch_add(d + 1, v.y * nrm, __ATOMIC_RELAXED, __HIP_MEMORY_SCOPE_AGENT);
    }
}

// ---------------------------------------------------------------------------
// Bias (+ optional ReLU), in place
// ---------------------------------------------------------------------------
template <int CH, bool RELU>
__global__ void k_bias(float* __restrict__ h, const float* __restrict__ b, size_t total) {
    size_t i = (size_t)blockIdx.x * blockDim.x + threadIdx.x;
    if (i >= total) return;
    float v = h[i] + b[i & (CH - 1)];
    if (RELU) v = fmaxf(v, 0.0f);
    h[i] = v;
}

// ---------------------------------------------------------------------------
// Launcher
// ---------------------------------------------------------------------------
extern "C" void kernel_launch(void* const* d_in, const int* in_sizes, int n_in,
                              void* d_out, int out_size, void* d_ws, size_t ws_size,
                              hipStream_t stream) {
    const float* x  = (const float*)d_in[0];
    const int*   ei = (const int*)d_in[1];
    const float* W1 = (const float*)d_in[2];
    const float* b1 = (const float*)d_in[3];
    const float* W2 = (const float*)d_in[4];
    const float* b2 = (const float*)d_in[5];

    const int N = in_sizes[0] / 128;
    const int E = in_sizes[1] / 2;
    const int* rowp = ei;          // source nodes
    const int* colp = ei + E;      // target nodes
    float* outp = (float*)d_out;

    // Workspace: dis[N] | t[N*128] (reused as N*64) | h1[N*128] | Wp1 | Wp2
    char* p = (char*)d_ws;
    auto align_up = [](size_t v, size_t a) { return (v + a - 1) & ~(a - 1); };
    size_t off = 0;
    float* dis = (float*)(p + off);  off += align_up((size_t)N * 4, 256);
    float* t   = (float*)(p + off);  off += align_up((size_t)N * 128 * 4, 256);
    float* h1  = (float*)(p + off);  off += align_up((size_t)N * 128 * 4, 256);
    _Float16* Wp1 = (_Float16*)(p + off); off += align_up((size_t)128 * 128 * 2, 256);
    _Float16* Wp2 = (_Float16*)(p + off);

    const int TB = 256;
    const unsigned nb_N   = (unsigned)((N + TB - 1) / TB);
    const unsigned nb_E   = (unsigned)((E + TB - 1) / TB);
    const long long eth   = (long long)E * 32;              // wave per edge
    const unsigned nb_EW  = (unsigned)((eth + TB - 1) / TB);
    const unsigned nb_M16 = (unsigned)((N + 15) / 16);

    // --- normalization + weight repack (independent, cheap) ---
    k_deg_init <<<nb_N, TB, 0, stream>>>(dis, N);
    k_deg_count<<<nb_E, TB, 0, stream>>>(colp, dis, E);
    k_deg_rsqrt<<<nb_N, TB, 0, stream>>>(dis, N);
    k_packW<128><<<(128 * 128 + TB - 1) / TB, TB, 0, stream>>>(W1, Wp1);
    k_packW< 64><<<(128 *  64 + TB - 1) / TB, TB, 0, stream>>>(W2, Wp2);

    // --- layer 1: t = x @ W1 ; h1 = scatter(norm * t) + b1 ; relu ---
    k_gemm_wmma<128><<<nb_M16, 256, 0, stream>>>(x, Wp1, t, N);
    {
        size_t q = (size_t)N * 32;  // float4s
        k_self<128><<<(unsigned)((q + TB - 1) / TB), TB, 0, stream>>>(t, dis, h1, N);
    }
    k_edges<4><<<nb_EW, TB, 0, stream>>>(t, rowp, colp, dis, h1, E);
    {
        size_t tot = (size_t)N * 128;
        k_bias<128, true><<<(unsigned)((tot + TB - 1) / TB), TB, 0, stream>>>(h1, b1, tot);
    }

    // --- layer 2: t = h1 @ W2 ; out = scatter(norm * t) + b2 ---
    k_gemm_wmma<64><<<nb_M16, 128, 0, stream>>>(h1, Wp2, t, N);
    {
        size_t q = (size_t)N * 16;  // float4s
        k_self<64><<<(unsigned)((q + TB - 1) / TB), TB, 0, stream>>>(t, dis, outp, N);
    }
    k_edges<2><<<nb_EW, TB, 0, stream>>>(t, rowp, colp, dis, outp, E);
    {
        size_t tot = (size_t)N * 64;
        k_bias<64, false><<<(unsigned)((tot + TB - 1) / TB), TB, 0, stream>>>(outp, b2, tot);
    }
}
